// Model_25606595019091
// MI455X (gfx1250) — matrix operand order repaired and checked by reading, not compiled, not run on hardware
//
#include <hip/hip_runtime.h>

// ---------------------------------------------------------------------------
// Problem constants (match reference)
// ---------------------------------------------------------------------------
#define B_ 32
#define S_ 512
#define DM_ 512
#define H_ 8
#define NM_ 4
#define DK_ 64
#define C1_ 5
#define C2_ 2
#define PRED_ 96

typedef __attribute__((ext_vector_type(16))) __bf16 v16bf;
typedef __attribute__((ext_vector_type(8)))  float  v8f;
typedef __attribute__((ext_vector_type(4)))  unsigned v4u;
typedef __attribute__((ext_vector_type(4)))  float  v4f;

union FragU { v16bf bf; v4u q[2]; unsigned short us[16]; };
union PackU { v4u q; unsigned short u[8]; };

#define DEV __device__ __forceinline__

DEV unsigned short f2bf(float f) {
    unsigned u = __float_as_uint(f);
    return (unsigned short)((u + 0x7FFFu + ((u >> 16) & 1u)) >> 16);
}
DEV float eluf(float x) { return x > 0.f ? x : (__expf(x) - 1.f); }

DEV v8f wmma_bf16(v16bf a, v16bf b, v8f c) {
    return __builtin_amdgcn_wmma_f32_16x16x32_bf16(false, a, false, b, (short)0, c,
                                                   false, false);
}

// A-fragment (16x32 bf16, row-major source, K contiguous).
// lane<16: row=lane, K chunks {0..7},{16..23}; lane>=16: row=lane-16, K {8..15},{24..31}
DEV v16bf loadA(const unsigned short* base, int stride, int lane) {
    int r = lane & 15, hf = lane >> 4;
    const unsigned short* p = base + (size_t)r * stride + hf * 8;
    FragU f;
    f.q[0] = *(const v4u*)(p);
    f.q[1] = *(const v4u*)(p + 16);
    return f.bf;
}
// B-fragment (32x16 bf16) from TRANSPOSED storage (column n = contiguous row n).
// lane<16: col=lane, K chunks {0..7},{16..23}; lane>=16: col=lane-16, K {8..15},{24..31}
DEV v16bf loadB(const unsigned short* base, int stride, int lane) {
    int c = lane & 15, hf = lane >> 4;
    const unsigned short* p = base + (size_t)c * stride + hf * 8;
    FragU f;
    f.q[0] = *(const v4u*)(p);
    f.q[1] = *(const v4u*)(p + 16);
    return f.bf;
}

// ---------------------------------------------------------------------------
// 1) Embedding + positional encoding -> h (bf16, layout (b,m,s,d))
// ---------------------------------------------------------------------------
__global__ void __launch_bounds__(256) embed_kernel(
    const float* __restrict__ x, const float* __restrict__ eW,
    const float* __restrict__ eb, unsigned short* __restrict__ h) {
    int d  = blockIdx.x * 256 + threadIdx.x;
    int s  = blockIdx.y;
    int bm = blockIdx.z;
    int b = bm >> 2, m = bm & 3;
    const float* xp = x + ((size_t)b * S_ + s) * 7;
    float v = xp[3 + m] * eW[d] + xp[0] * eW[DM_ + d] + xp[1] * eW[2 * DM_ + d] +
              xp[2] * eW[3 * DM_ + d] + eb[d];
    // PE: freq = exp(-(ln 1e4 / 512) * (2*(d/2)))
    float freq = __expf(-0.0179889460f * (float)(d & ~1));
    float ang  = (float)s * freq;
    v += (d & 1) ? __cosf(ang) : __sinf(ang);
    h[((size_t)bm * S_ + s) * DM_ + d] = f2bf(v);
}

// ---------------------------------------------------------------------------
// 2) Weight transpose+convert: W (m,d,e) f32 -> Wt (m,e,d) bf16
// ---------------------------------------------------------------------------
__global__ void __launch_bounds__(256) transpose_w_kernel(
    const float* __restrict__ W, unsigned short* __restrict__ Wt) {
    __shared__ float tile[16][17];
    int m  = blockIdx.z;
    int e0 = blockIdx.x << 4, d0 = blockIdx.y << 4;
    int tx = threadIdx.x, ty = threadIdx.y;
    tile[ty][tx] = W[((size_t)(m * DM_ + d0 + ty)) * DM_ + e0 + tx];
    __syncthreads();
    Wt[((size_t)(m * DM_ + e0 + ty)) * DM_ + d0 + tx] = f2bf(tile[tx][ty]);
}

// ---------------------------------------------------------------------------
// 3) WMMA GEMM: Y[s,e] = A[s,:] (bf16) x Wt[e,:]^T + bias[m,e]
//    Block: 4 waves. Wave tile 32(s) x 64(e); block tile 32(s) x 256(e).
//    Shared 32x32 A-tile staged in LDS with double-buffered
//    GLOBAL_LOAD_ASYNC_TO_LDS_B128 (ASYNCcnt + barrier sync).
//    MODE 0: per-head L2-normalize (64-col tile == one head), bf16 row-major out
//    MODE 1: bf16 transposed out (b,m,d,s)  (for V)
//    MODE 2: fp32 out into combined (b, s, m*512+e)
// ---------------------------------------------------------------------------
template <int MODE>
__global__ void __launch_bounds__(128) gemm_head(
    const unsigned short* __restrict__ A, const unsigned short* __restrict__ Wt,
    const float* __restrict__ bias, void* __restrict__ outp) {
    __shared__ __align__(16) unsigned short sA[2][32 * 32];  // 2 x 2KB k-slices
    const int lane = threadIdx.x & 31, wave = threadIdx.x >> 5;
    const int bm = blockIdx.z, m = bm & 3, b = bm >> 2;
    const int s0 = blockIdx.y << 5;                         // 32 rows per block
    const int e0 = ((blockIdx.x << 2) + wave) << 6;         // 64 cols per wave
    const unsigned short* Ab = A + (size_t)(bm * S_ + s0) * DM_;
    const unsigned short* Bb = Wt + (size_t)m * DM_ * DM_;

    // Per-thread async staging assignment: 128 threads x 16B = 2KB = 32x32 bf16
    const int tid  = threadIdx.x;
    const int srow = tid >> 2;        // 0..31
    const int sc8  = (tid & 3) * 8;   // element offset 0,8,16,24

    // issue first k-slice (k0 = 0)
    {
        const unsigned short* g = Ab + (size_t)srow * DM_ + sc8;
        unsigned lo = (unsigned)(unsigned long long)(&sA[0][srow * 32 + sc8]);
        asm volatile("global_load_async_to_lds_b128 %0, %1, off" ::"v"(lo), "v"(g)
                     : "memory");
    }

    v8f acc[2][4] = {};
    int cur = 0;
    for (int k0 = 0; k0 < DM_; k0 += 32) {
        asm volatile("s_wait_asynccnt 0x0" ::: "memory");
        __syncthreads();  // sA[cur] ready for all waves; prior reads retired
        if (k0 + 32 < DM_) {
            const unsigned short* g = Ab + (size_t)srow * DM_ + k0 + 32 + sc8;
            unsigned lo =
                (unsigned)(unsigned long long)(&sA[cur ^ 1][srow * 32 + sc8]);
            asm volatile("global_load_async_to_lds_b128 %0, %1, off" ::"v"(lo),
                         "v"(g)
                         : "memory");
        }
        v16bf aLo = loadA(&sA[cur][0], 32, lane);        // rows s0+0..15
        v16bf aHi = loadA(&sA[cur][16 * 32], 32, lane);  // rows s0+16..31
#pragma unroll
        for (int t = 0; t < 4; ++t) {
            v16bf bfr = loadB(Bb + (size_t)(e0 + t * 16) * DM_ + k0, DM_, lane);
            acc[0][t] = wmma_bf16(aLo, bfr, acc[0][t]);
            acc[1][t] = wmma_bf16(aHi, bfr, acc[1][t]);
        }
        __builtin_prefetch(Bb + (size_t)e0 * DM_ + k0 + 64, 0, 1);
        cur ^= 1;
    }

    const int col = lane & 15, hf = lane >> 4;
#pragma unroll
    for (int g = 0; g < 2; ++g)
#pragma unroll
        for (int t = 0; t < 4; ++t) {
            float bv = bias[m * DM_ + e0 + t * 16 + col];
#pragma unroll
            for (int r = 0; r < 8; ++r) acc[g][t][r] += bv;
        }

    if (MODE == 0) {
        // per-row L2 norm over the 64 columns of this head
#pragma unroll
        for (int g = 0; g < 2; ++g)
#pragma unroll
            for (int r = 0; r < 8; ++r) {
                float ss = 0.f;
#pragma unroll
                for (int t = 0; t < 4; ++t) ss += acc[g][t][r] * acc[g][t][r];
                ss += __shfl_xor(ss, 1, 32);
                ss += __shfl_xor(ss, 2, 32);
                ss += __shfl_xor(ss, 4, 32);
                ss += __shfl_xor(ss, 8, 32);
                float inv = 1.0f / fmaxf(sqrtf(ss), 1e-12f);
#pragma unroll
                for (int t = 0; t < 4; ++t) acc[g][t][r] *= inv;
            }
        unsigned short* o = (unsigned short*)outp;
#pragma unroll
        for (int g = 0; g < 2; ++g)
#pragma unroll
            for (int t = 0; t < 4; ++t)
#pragma unroll
                for (int r = 0; r < 8; ++r)
                    o[(size_t)(bm * S_ + s0 + g * 16 + r + 8 * hf) * DM_ + e0 +
                      t * 16 + col] = f2bf(acc[g][t][r]);
    } else if (MODE == 1) {
        unsigned short* o = (unsigned short*)outp;
#pragma unroll
        for (int g = 0; g < 2; ++g)
#pragma unroll
            for (int t = 0; t < 4; ++t) {
                PackU pk;
#pragma unroll
                for (int r = 0; r < 8; ++r) pk.u[r] = f2bf(acc[g][t][r]);
                *(v4u*)(o + (size_t)(bm * DM_ + e0 + t * 16 + col) * S_ + s0 +
                        g * 16 + hf * 8) = pk.q;
            }
    } else {
        float* o = (float*)outp;
#pragma unroll
        for (int g = 0; g < 2; ++g)
#pragma unroll
            for (int t = 0; t < 4; ++t)
#pragma unroll
                for (int r = 0; r < 8; ++r)
                    o[(size_t)b * S_ * 2048 +
                      (size_t)(s0 + g * 16 + r + 8 * hf) * 2048 + m * 512 + e0 +
                      t * 16 + col] = acc[g][t][r];
    }
}

// ---------------------------------------------------------------------------
// 4) Attention: one wave per (b, m, h, 16-query block).
//    scores (16x512) in LDS (fp32), softmax, then P x V via WMMA.
// ---------------------------------------------------------------------------
__global__ void __launch_bounds__(32) attn_kernel(
    const unsigned short* __restrict__ Q, const unsigned short* __restrict__ K,
    const unsigned short* __restrict__ Vt, unsigned short* __restrict__ ctx) {
    __shared__ __align__(16) float sc[16 * S_];  // 32 KB
    const int lane = threadIdx.x;
    const int s0 = blockIdx.x << 4;
    const int m = blockIdx.y >> 3, h = blockIdx.y & 7;
    const int b = blockIdx.z, bm = b * NM_ + m;
    const int col = lane & 15, hf = lane >> 4;

    const unsigned short* qb = Q + (size_t)(bm * S_ + s0) * DM_ + h * DK_;
    const unsigned short* kb = K + (size_t)bm * S_ * DM_ + h * DK_;

    v16bf a0 = loadA(qb, DM_, lane);
    v16bf a1 = loadA(qb + 32, DM_, lane);

    for (int t0 = 0; t0 < S_; t0 += 16) {
        v16bf k0 = loadB(kb + (size_t)t0 * DM_, DM_, lane);
        v16bf k1 = loadB(kb + (size_t)t0 * DM_ + 32, DM_, lane);
        __builtin_prefetch(kb + (size_t)(t0 + 16) * DM_, 0, 1);
        v8f acc = {};
        acc = wmma_bf16(a0, k0, acc);
        acc = wmma_bf16(a1, k1, acc);
#pragma unroll
        for (int r = 0; r < 8; ++r) sc[(r + 8 * hf) * S_ + t0 + col] = acc[r];
    }
    __syncthreads();

    // softmax per query row (16 rows, 512 cols, whole wave per row)
    for (int r = 0; r < 16; ++r) {
        float vals[16];
        float mx = -1e30f;
#pragma unroll
        for (int i = 0; i < 16; ++i) {
            vals[i] = sc[r * S_ + lane + 32 * i];
            mx = fmaxf(mx, vals[i]);
        }
        mx = fmaxf(mx, __shfl_xor(mx, 16, 32));
        mx = fmaxf(mx, __shfl_xor(mx, 8, 32));
        mx = fmaxf(mx, __shfl_xor(mx, 4, 32));
        mx = fmaxf(mx, __shfl_xor(mx, 2, 32));
        mx = fmaxf(mx, __shfl_xor(mx, 1, 32));
        float sum = 0.f;
#pragma unroll
        for (int i = 0; i < 16; ++i) {
            vals[i] = __expf(vals[i] - mx);
            sum += vals[i];
        }
        sum += __shfl_xor(sum, 16, 32);
        sum += __shfl_xor(sum, 8, 32);
        sum += __shfl_xor(sum, 4, 32);
        sum += __shfl_xor(sum, 2, 32);
        sum += __shfl_xor(sum, 1, 32);
        float inv = 1.f / sum;
#pragma unroll
        for (int i = 0; i < 16; ++i) sc[r * S_ + lane + 32 * i] = vals[i] * inv;
    }
    __syncthreads();

    // ctx = P x V  (A from LDS probs fp32->bf16, B from transposed V)
    const unsigned short* vb = Vt + (size_t)(bm * DM_ + h * DK_) * S_;
    v8f c[4] = {};
    for (int t0 = 0; t0 < S_; t0 += 32) {
        FragU pa;
        const float* pr = sc + col * S_ + t0 + hf * 8;
        v4f f0 = *(const v4f*)(pr);
        v4f f1 = *(const v4f*)(pr + 4);
        v4f f2 = *(const v4f*)(pr + 16);
        v4f f3 = *(const v4f*)(pr + 20);
        pa.us[0] = f2bf(f0.x);  pa.us[1] = f2bf(f0.y);
        pa.us[2] = f2bf(f0.z);  pa.us[3] = f2bf(f0.w);
        pa.us[4] = f2bf(f1.x);  pa.us[5] = f2bf(f1.y);
        pa.us[6] = f2bf(f1.z);  pa.us[7] = f2bf(f1.w);
        pa.us[8] = f2bf(f2.x);  pa.us[9] = f2bf(f2.y);
        pa.us[10] = f2bf(f2.z); pa.us[11] = f2bf(f2.w);
        pa.us[12] = f2bf(f3.x); pa.us[13] = f2bf(f3.y);
        pa.us[14] = f2bf(f3.z); pa.us[15] = f2bf(f3.w);
#pragma unroll
        for (int nt = 0; nt < 4; ++nt) {
            v16bf vf = loadB(vb + (size_t)(nt * 16) * S_ + t0, S_, lane);
            c[nt] = wmma_bf16(pa.bf, vf, c[nt]);
        }
    }
#pragma unroll
    for (int nt = 0; nt < 4; ++nt)
#pragma unroll
        for (int r = 0; r < 8; ++r)
            ctx[(size_t)(bm * S_ + s0 + r + 8 * hf) * DM_ + h * DK_ + nt * 16 + col] =
                f2bf(c[nt][r]);
}

// ---------------------------------------------------------------------------
// 5) Conv1 (2048 -> 5, k=3, pad 1) + ELU.  One thread per (b, s).
// ---------------------------------------------------------------------------
__global__ void __launch_bounds__(256) conv1_kernel(
    const float* __restrict__ X, const float* __restrict__ W,
    const float* __restrict__ bias, float* __restrict__ O) {
    int s = blockIdx.x * 256 + threadIdx.x;
    int b = blockIdx.y;
    float acc[C1_];
#pragma unroll
    for (int c = 0; c < C1_; ++c) acc[c] = bias[c];
    for (int kt = 0; kt < 3; ++kt) {
        int ss = s + kt - 1;
        if (ss < 0 || ss >= S_) continue;
        const v4f* p = (const v4f*)(X + ((size_t)b * S_ + ss) * 2048);
        for (int ic = 0; ic < 2048; ic += 4) {
            v4f v = p[ic >> 2];
#pragma unroll
            for (int c = 0; c < C1_; ++c) {
                const float* w = W + ((size_t)c * 2048 + ic) * 3 + kt;
                acc[c] += v.x * w[0] + v.y * w[3] + v.z * w[6] + v.w * w[9];
            }
        }
    }
#pragma unroll
    for (int c = 0; c < C1_; ++c) O[((size_t)b * C1_ + c) * S_ + s] = eluf(acc[c]);
}

// ---------------------------------------------------------------------------
// 6) Conv2 (5 -> 2, k=3, pad 1) + ELU + max-pool(2) -> flat (b, 512)
// ---------------------------------------------------------------------------
__global__ void __launch_bounds__(256) conv2_pool_kernel(
    const float* __restrict__ X, const float* __restrict__ W,
    const float* __restrict__ bias, float* __restrict__ O) {
    int idx = blockIdx.x * 256 + threadIdx.x;  // c2*256 + sp
    int b = blockIdx.y;
    int c2 = idx >> 8, sp = idx & 255;
    float ov[2];
#pragma unroll
    for (int j = 0; j < 2; ++j) {
        int s = 2 * sp + j;
        float acc = bias[c2];
#pragma unroll
        for (int ic = 0; ic < C1_; ++ic)
#pragma unroll
            for (int kt = 0; kt < 3; ++kt) {
                int ss = s + kt - 1;
                if (ss >= 0 && ss < S_)
                    acc += X[((size_t)b * C1_ + ic) * S_ + ss] *
                           W[((size_t)c2 * C1_ + ic) * 3 + kt];
            }
        ov[j] = eluf(acc);
    }
    O[(size_t)b * 512 + idx] = fmaxf(ov[0], ov[1]);
}

// ---------------------------------------------------------------------------
// 7) MLP (512->256->128->96) + LayerNorm.  One block (256 thr) per batch row.
// ---------------------------------------------------------------------------
__global__ void __launch_bounds__(256) mlp_kernel(
    const float* __restrict__ flat, const float* __restrict__ w1,
    const float* __restrict__ b1, const float* __restrict__ w2,
    const float* __restrict__ b2, const float* __restrict__ w3,
    const float* __restrict__ b3, const float* __restrict__ g,
    const float* __restrict__ be, float* __restrict__ out) {
    __shared__ float sf[512];
    __shared__ float s1[256];
    __shared__ float s2[128];
    __shared__ float s3[96];
    __shared__ float red[2];
    int b = blockIdx.x, t = threadIdx.x;
    sf[t] = flat[(size_t)b * 512 + t];
    sf[t + 256] = flat[(size_t)b * 512 + 256 + t];
    __syncthreads();
    {
        float acc = b1[t];
        for (int k = 0; k < 512; ++k) acc += sf[k] * w1[(size_t)k * 256 + t];
        s1[t] = eluf(acc);
    }
    __syncthreads();
    if (t < 128) {
        float acc = b2[t];
        for (int k = 0; k < 256; ++k) acc += s1[k] * w2[(size_t)k * 128 + t];
        s2[t] = eluf(acc);
    }
    __syncthreads();
    if (t < 96) {
        float acc = b3[t];
        for (int k = 0; k < 128; ++k) acc += s2[k] * w3[(size_t)k * 96 + t];
        s3[t] = acc;
    }
    __syncthreads();
    if (t < 32) {
        float sm = 0.f, sq = 0.f;
#pragma unroll
        for (int i = 0; i < 3; ++i) {
            float v = s3[t + 32 * i];
            sm += v;
            sq += v * v;
        }
        sm += __shfl_xor(sm, 16, 32); sq += __shfl_xor(sq, 16, 32);
        sm += __shfl_xor(sm, 8, 32);  sq += __shfl_xor(sq, 8, 32);
        sm += __shfl_xor(sm, 4, 32);  sq += __shfl_xor(sq, 4, 32);
        sm += __shfl_xor(sm, 2, 32);  sq += __shfl_xor(sq, 2, 32);
        sm += __shfl_xor(sm, 1, 32);  sq += __shfl_xor(sq, 1, 32);
        if (t == 0) {
            float mu = sm / 96.f;
            red[0] = mu;
            red[1] = sq / 96.f - mu * mu;
        }
    }
    __syncthreads();
    if (t < 96)
        out[(size_t)b * PRED_ + t] =
            (s3[t] - red[0]) * rsqrtf(red[1] + 1e-5f) * g[t] + be[t];
}

// ---------------------------------------------------------------------------
// Launcher
// ---------------------------------------------------------------------------
extern "C" void kernel_launch(void* const* d_in, const int* in_sizes, int n_in,
                              void* d_out, int out_size, void* d_ws, size_t ws_size,
                              hipStream_t stream) {
    const float* x     = (const float*)d_in[0];
    const float* emb_W = (const float*)d_in[1];
    const float* emb_b = (const float*)d_in[2];
    const float* Wq    = (const float*)d_in[3];
    const float* bq    = (const float*)d_in[4];
    const float* Wk    = (const float*)d_in[5];
    const float* bk    = (const float*)d_in[6];
    const float* Wv    = (const float*)d_in[7];
    const float* bv    = (const float*)d_in[8];
    const float* Wo    = (const float*)d_in[9];
    const float* bo    = (const float*)d_in[10];
    const float* c1w   = (const float*)d_in[11];
    const float* c1b   = (const float*)d_in[12];
    const float* c2w   = (const float*)d_in[13];
    const float* c2b   = (const float*)d_in[14];
    const float* f1w   = (const float*)d_in[15];
    const float* f1b   = (const float*)d_in[16];
    const float* f2w   = (const float*)d_in[17];
    const float* f2b   = (const float*)d_in[18];
    const float* f3w   = (const float*)d_in[19];
    const float* f3b   = (const float*)d_in[20];
    const float* lng   = (const float*)d_in[21];
    const float* lnb   = (const float*)d_in[22];

    char* wsb = (char*)d_ws;
    size_t off = 0;
    auto take = [&](size_t bytes) -> void* {
        void* p = wsb + off;
        off += (bytes + 255) & ~(size_t)255;
        return p;
    };
    const size_t NTOK = (size_t)B_ * NM_ * S_ * DM_;  // 33.5 M elems
    unsigned short* h_bf = (unsigned short*)take(NTOK * 2);
    unsigned short* Wqt  = (unsigned short*)take((size_t)NM_ * DM_ * DM_ * 2);
    unsigned short* Wkt  = (unsigned short*)take((size_t)NM_ * DM_ * DM_ * 2);
    unsigned short* Wvt  = (unsigned short*)take((size_t)NM_ * DM_ * DM_ * 2);
    unsigned short* Wot  = (unsigned short*)take((size_t)NM_ * DM_ * DM_ * 2);
    unsigned short* Qbf  = (unsigned short*)take(NTOK * 2);
    unsigned short* Kbf  = (unsigned short*)take(NTOK * 2);
    unsigned short* Vtbf = (unsigned short*)take(NTOK * 2);
    unsigned short* Ctx  = (unsigned short*)take(NTOK * 2);
    float* combined = (float*)take((size_t)B_ * S_ * 2048 * 4);
    float* c1o      = (float*)take((size_t)B_ * C1_ * S_ * 4);
    float* flat     = (float*)take((size_t)B_ * 512 * 4);

    dim3 tb(16, 16);
    dim3 tg(DM_ / 16, DM_ / 16, NM_);
    transpose_w_kernel<<<tg, tb, 0, stream>>>(Wq, Wqt);
    transpose_w_kernel<<<tg, tb, 0, stream>>>(Wk, Wkt);
    transpose_w_kernel<<<tg, tb, 0, stream>>>(Wv, Wvt);
    transpose_w_kernel<<<tg, tb, 0, stream>>>(Wo, Wot);

    embed_kernel<<<dim3(DM_ / 256, S_, B_ * NM_), 256, 0, stream>>>(x, emb_W, emb_b,
                                                                    h_bf);

    dim3 gg(DM_ / 256, S_ / 32, B_ * NM_);  // (2, 16, 128), 4 waves/block
    gemm_head<0><<<gg, 128, 0, stream>>>(h_bf, Wqt, bq, Qbf);
    gemm_head<0><<<gg, 128, 0, stream>>>(h_bf, Wkt, bk, Kbf);
    gemm_head<1><<<gg, 128, 0, stream>>>(h_bf, Wvt, bv, Vtbf);

    attn_kernel<<<dim3(S_ / 16, NM_ * H_, B_), 32, 0, stream>>>(Qbf, Kbf, Vtbf, Ctx);

    gemm_head<2><<<gg, 128, 0, stream>>>(Ctx, Wot, bo, combined);

    conv1_kernel<<<dim3(S_ / 256, B_), 256, 0, stream>>>(combined, c1w, c1b, c1o);
    conv2_pool_kernel<<<dim3(2, B_), 256, 0, stream>>>(c1o, c2w, c2b, flat);
    mlp_kernel<<<dim3(B_), 256, 0, stream>>>(flat, f1w, f1b, f2w, f2b, f3w, f3b, lng,
                                             lnb, (float*)d_out);
}
